// ToMeBlock_35450660061685
// MI455X (gfx1250) — matrix operand order repaired and checked by reading, not compiled, not run on hardware
//
#include <hip/hip_runtime.h>
#include <hip/hip_bf16.h>
#include <math.h>

// ---------------- types ----------------
typedef __attribute__((ext_vector_type(16))) __bf16 v16bf;
typedef __attribute__((ext_vector_type(8)))  __bf16 v8bf;
typedef __attribute__((ext_vector_type(8)))  float  v8f;
typedef __attribute__((ext_vector_type(4)))  unsigned int u32x4;

// ---------------- constants ----------------
#define CDIM   768
#define HEADS  12
#define DH     64
#define SEQ    577
#define BATCH  32
#define MROWS  (BATCH*SEQ)      // 18464
#define MPAD   (145*128)        // 18560: padded row count for GEMM A-side tiles
#define HID    3072
#define SPAD   640              // padded sequence for q/k/v buffers (40*16)
#define NIMG   576
#define RKEEP  288

static __device__ __forceinline__ __bf16 f2bf(float f) {
    union { float f; unsigned int u; } a; a.f = f;
    unsigned int r = a.u + 0x7FFFu + ((a.u >> 16) & 1u);
    union { unsigned short s; __bf16 b; } c; c.s = (unsigned short)(r >> 16);
    return c.b;
}

static __device__ __forceinline__ v8f wmma_bf16(v16bf a, v16bf b, v8f c) {
    return __builtin_amdgcn_wmma_f32_16x16x32_bf16(false, a, false, b, (short)0, c, false, false);
}

// combine two aligned 8-element bf16 chunks into one 16-element A/B fragment
static __device__ __forceinline__ v16bf frag16(const __bf16* p0, const __bf16* p1) {
    v8bf a = *(const v8bf*)p0;
    v8bf b = *(const v8bf*)p1;
    v16bf r;
#pragma unroll
    for (int i = 0; i < 8; ++i) { r[i] = a[i]; r[i + 8] = b[i]; }
    return r;
}

// CDNA5 async copy: 16 bytes global -> LDS, tracked by ASYNCcnt
static __device__ __forceinline__ void async_g2l_b128(const void* gaddr, void* lds, int off) {
    unsigned la = (unsigned)(size_t)lds;
    unsigned long long ga = (unsigned long long)(size_t)gaddr;
    if (off == 0)
        asm volatile("global_load_async_to_lds_b128 %0, %1, off" :: "v"(la), "v"(ga) : "memory");
    else
        asm volatile("global_load_async_to_lds_b128 %0, %1, off offset:16" :: "v"(la), "v"(ga) : "memory");
}
static __device__ __forceinline__ void wait_async0() {
    asm volatile("s_wait_asynccnt 0x0" ::: "memory");
}
static __device__ __forceinline__ void wait_ds0() {
    asm volatile("s_wait_dscnt 0x0" ::: "memory");
}

// ---------------- elementwise cast (weights fp32 -> bf16) ----------------
__global__ __launch_bounds__(256) void cast_bf16_kernel(const float* __restrict__ in,
                                                        __bf16* __restrict__ out, long n) {
    long i = (long)blockIdx.x * 256 + threadIdx.x;
    if (i < n) out[i] = f2bf(in[i]);
}

// ---------------- LayerNorm (row of 768) -> bf16 ----------------
__global__ __launch_bounds__(256) void ln_bf16_kernel(const float* __restrict__ x,
                                                      const float* __restrict__ g,
                                                      const float* __restrict__ b,
                                                      __bf16* __restrict__ out, int rows) {
    __shared__ float red[256];
    int row = blockIdx.x;
    if (row >= rows) return;
    int tid = threadIdx.x;
    const float* xr = x + (long)row * CDIM;
    float v[3];
    float s = 0.f;
#pragma unroll
    for (int e = 0; e < 3; ++e) { v[e] = xr[tid + 256 * e]; s += v[e]; }
    red[tid] = s; __syncthreads();
    for (int off = 128; off > 0; off >>= 1) { if (tid < off) red[tid] += red[tid + off]; __syncthreads(); }
    float mu = red[0] * (1.0f / CDIM);
    __syncthreads();
    float d2 = 0.f;
#pragma unroll
    for (int e = 0; e < 3; ++e) { float d = v[e] - mu; d2 += d * d; }
    red[tid] = d2; __syncthreads();
    for (int off = 128; off > 0; off >>= 1) { if (tid < off) red[tid] += red[tid + off]; __syncthreads(); }
    float rs = rsqrtf(red[0] * (1.0f / CDIM) + 1e-5f);
    __bf16* orow = out + (long)row * CDIM;
#pragma unroll
    for (int e = 0; e < 3; ++e) {
        int c = tid + 256 * e;
        orow[c] = f2bf((v[e] - mu) * rs * g[c] + b[c]);
    }
}

// ---------------- generic WMMA GEMM:  C[M,N] = A[M,K] * B[N,K]^T (+bias, epilogue) ----------------
// A must have >= gridDim.y*128 allocated rows; B >= gridDim.x*128 rows (loads unguarded,
// fed by global_load_async_to_lds_b128; bad rows feed only discarded output tiles).
enum { EPI_QKV = 0, EPI_RES = 1, EPI_GELU = 2, EPI_SIM = 3 };

template <int EPI>
__global__ __launch_bounds__(256, 1)
void gemm_nt_kernel(const __bf16* __restrict__ A, const __bf16* __restrict__ Bw,
                    const float* __restrict__ bias,
                    int M, int N, int K,
                    long asb, long bsb, long csb,           // batch strides (elements)
                    float* __restrict__ outF, __bf16* __restrict__ outBF,
                    const float* __restrict__ res,
                    __bf16* __restrict__ qo, __bf16* __restrict__ ko, __bf16* __restrict__ vo) {
    const int bz = blockIdx.z;
    A  += (long)bz * asb;
    Bw += (long)bz * bsb;

    __shared__ __align__(16) __bf16 As[128][40];
    __shared__ __align__(16) __bf16 Bs[128][40];

    const int tid  = threadIdx.x;
    const int lane = tid & 31;
    const int wid  = tid >> 5;
    const int wm   = wid & 3;         // 4 waves along M
    const int wn   = wid >> 2;        // 2 waves along N
    const int hf   = lane >> 4;
    const int l16  = lane & 15;
    const int bm   = blockIdx.y * 128;
    const int bn   = blockIdx.x * 128;

    v8f acc[2][4];
#pragma unroll
    for (int i = 0; i < 2; ++i)
#pragma unroll
        for (int j = 0; j < 4; ++j) acc[i][j] = (v8f){0.f,0.f,0.f,0.f,0.f,0.f,0.f,0.f};

    const int ldrow = tid >> 1;
    const int ldcol = (tid & 1) * 16;
    const __bf16* gA = A + (long)(bm + ldrow) * K + ldcol;
    const __bf16* gB = Bw + (long)(bn + ldrow) * K + ldcol;
    __bf16* lA = &As[ldrow][ldcol];
    __bf16* lB = &Bs[ldrow][ldcol];

    for (int k0 = 0; k0 < K; k0 += 32) {
        // async tile copy: 32 bytes per thread per tile, straight into LDS
        async_g2l_b128(gA + k0, lA, 0);
        async_g2l_b128(gA + k0, lA, 16);
        async_g2l_b128(gB + k0, lB, 0);
        async_g2l_b128(gB + k0, lB, 16);
        if (k0 + 32 < K) __builtin_prefetch(gA + k0 + 32, 0, 3);
        wait_async0();
        __syncthreads();

        v16bf af[2], bf[4];
#pragma unroll
        for (int i = 0; i < 2; ++i) {
            int r = wm * 32 + i * 16 + l16;
            af[i] = frag16(&As[r][8 * hf], &As[r][16 + 8 * hf]);
        }
#pragma unroll
        for (int j = 0; j < 4; ++j) {
            int r = wn * 64 + j * 16 + l16;
            bf[j] = frag16(&Bs[r][8 * hf], &Bs[r][16 + 8 * hf]);
        }
#pragma unroll
        for (int i = 0; i < 2; ++i)
#pragma unroll
            for (int j = 0; j < 4; ++j)
                acc[i][j] = wmma_bf16(af[i], bf[j], acc[i][j]);
        __syncthreads();
    }

    // epilogue: C layout -> lane l16 = column, VGPR r -> row (r + 8*hf)
#pragma unroll
    for (int i = 0; i < 2; ++i)
#pragma unroll
        for (int j = 0; j < 4; ++j)
#pragma unroll
            for (int r = 0; r < 8; ++r) {
                int gm = bm + wm * 32 + i * 16 + r + 8 * hf;
                int gn = bn + wn * 64 + j * 16 + l16;
                if (gm >= M || gn >= N) continue;
                float val = acc[i][j][r];
                if (EPI != EPI_SIM) val += bias[gn];
                if (EPI == EPI_QKV) {
                    int part = gn / CDIM;
                    int rem  = gn - part * CDIM;
                    int hh   = rem >> 6;
                    int d    = rem & 63;
                    int bb   = gm / SEQ;
                    int ss   = gm - bb * SEQ;
                    long bhid = (long)(bb * HEADS + hh);
                    if (part == 0)      qo[(bhid * SPAD + ss) * DH + d] = f2bf(val);
                    else if (part == 1) ko[(bhid * SPAD + ss) * DH + d] = f2bf(val);
                    else                vo[(bhid * DH + d) * SPAD + ss] = f2bf(val); // V transposed
                } else if (EPI == EPI_RES) {
                    long idx = (long)gm * N + gn;
                    outF[idx] = val + res[idx];
                } else if (EPI == EPI_GELU) {
                    float gl = 0.5f * val * (1.0f + erff(val * 0.70710678f));
                    outBF[(long)gm * N + gn] = f2bf(gl);
                } else { // EPI_SIM
                    float s = (gm == gn) ? -1.0f : val;
                    outF[(long)bz * csb + (long)gm * N + gn] = s;
                }
            }
}

// ---------------- fused flash attention (dh=64), K row-major, V pre-transposed ----------------
__global__ __launch_bounds__(128, 1)
void attn_flash_kernel(const __bf16* __restrict__ qb, const __bf16* __restrict__ kb,
                       const __bf16* __restrict__ vtb, __bf16* __restrict__ attn) {
    const int b = blockIdx.z, h = blockIdx.y, qblk = blockIdx.x;
    const int tid = threadIdx.x, lane = tid & 31, w = tid >> 5;
    const int hf = lane >> 4, l16 = lane & 15;

    __shared__ __align__(16) __bf16 Ps[4][16][40];   // per-wave P transpose staging

    const long bh = (long)b * HEADS + h;
    const __bf16* qbase  = qb  + bh * SPAD * DH;
    const __bf16* kbase  = kb  + bh * SPAD * DH;
    const __bf16* vtbase = vtb + bh * DH * SPAD;     // [64][SPAD]

    // Q fragments (16 rows per wave, dh split into two 32-wide K steps)
    const int qrow = qblk * 64 + w * 16 + l16;
    const __bf16* qp = qbase + (long)qrow * DH;
    v16bf qf0 = frag16(qp + 8 * hf,      qp + 16 + 8 * hf);
    v16bf qf1 = frag16(qp + 32 + 8 * hf, qp + 48 + 8 * hf);

    v8f o[4];
#pragma unroll
    for (int j = 0; j < 4; ++j) o[j] = (v8f){0.f,0.f,0.f,0.f,0.f,0.f,0.f,0.f};
    float mx[8], ls[8];
#pragma unroll
    for (int r = 0; r < 8; ++r) { mx[r] = -1e30f; ls[r] = 0.f; }

    const int nkb = (SEQ + 31) / 32; // 19
    for (int kbk = 0; kbk < nkb; ++kbk) {
        // scores: 16 q-rows x 32 keys per wave (B fragments straight from global K rows)
        v8f sa[2];
#pragma unroll
        for (int j = 0; j < 2; ++j) {
            sa[j] = (v8f){0.f,0.f,0.f,0.f,0.f,0.f,0.f,0.f};
            int key = kbk * 32 + j * 16 + l16;
            const __bf16* kp = kbase + (long)key * DH;
            v16bf b0 = frag16(kp + 8 * hf,      kp + 16 + 8 * hf);
            v16bf b1 = frag16(kp + 32 + 8 * hf, kp + 48 + 8 * hf);
            sa[j] = wmma_bf16(qf0, b0, sa[j]);
            sa[j] = wmma_bf16(qf1, b1, sa[j]);
        }

        // online softmax (row stats live in the 16-lane half holding that row)
        float corr[8];
        int k0i = kbk * 32 + l16;
        int k1i = kbk * 32 + 16 + l16;
#pragma unroll
        for (int r = 0; r < 8; ++r) {
            float v0 = sa[0][r] * 0.125f;
            float v1 = sa[1][r] * 0.125f;
            if (k0i >= SEQ) v0 = -1e30f;
            if (k1i >= SEQ) v1 = -1e30f;
            float tm = fmaxf(v0, v1);
            tm = fmaxf(tm, __shfl_xor(tm, 1));
            tm = fmaxf(tm, __shfl_xor(tm, 2));
            tm = fmaxf(tm, __shfl_xor(tm, 4));
            tm = fmaxf(tm, __shfl_xor(tm, 8));
            float nm = fmaxf(mx[r], tm);
            corr[r] = __expf(mx[r] - nm);
            float p0 = __expf(v0 - nm);
            float p1 = __expf(v1 - nm);
            float ps = p0 + p1;
            ps += __shfl_xor(ps, 1);
            ps += __shfl_xor(ps, 2);
            ps += __shfl_xor(ps, 4);
            ps += __shfl_xor(ps, 8);
            ls[r] = ls[r] * corr[r] + ps;
            mx[r] = nm;
            int m = r + 8 * hf;
            Ps[w][m][l16]      = f2bf(p0);   // transpose P through per-wave LDS
            Ps[w][m][16 + l16] = f2bf(p1);
        }
#pragma unroll
        for (int j = 0; j < 4; ++j)
#pragma unroll
            for (int r = 0; r < 8; ++r) o[j][r] *= corr[r];

        wait_ds0();  // wave-private region: DS ops in-order, make stores visible to loads

        // P @ V: A = transposed P, B fragments contiguous from pre-transposed V
        v16bf pa = frag16(&Ps[w][l16][8 * hf], &Ps[w][l16][16 + 8 * hf]);
#pragma unroll
        for (int j = 0; j < 4; ++j) {
            int d = j * 16 + l16;
            const __bf16* vp = vtbase + (long)d * SPAD + kbk * 32;
            v16bf vf = frag16(vp + 8 * hf, vp + 16 + 8 * hf);
            o[j] = wmma_bf16(pa, vf, o[j]);
        }
    }

    // write out (bf16, [B*S][768] with head interleave)
#pragma unroll
    for (int j = 0; j < 4; ++j)
#pragma unroll
        for (int r = 0; r < 8; ++r) {
            int srow = qblk * 64 + w * 16 + r + 8 * hf;
            if (srow < SEQ) {
                float val = o[j][r] / fmaxf(ls[r], 1e-20f);
                attn[((long)b * SEQ + srow) * CDIM + h * DH + j * 16 + l16] = f2bf(val);
            }
        }
}

// ---------------- ToMe helpers ----------------
__global__ __launch_bounds__(256) void norm_rows_kernel(const float* __restrict__ x2,
                                                        __bf16* __restrict__ xn) {
    __shared__ float red[256];
    int row = blockIdx.x;                 // 0 .. B*576-1
    int b = row / NIMG, t = row - b * NIMG;
    int tid = threadIdx.x;
    const float* src = x2 + ((long)b * SEQ + 1 + t) * CDIM;
    float v[3];
    float s = 0.f;
#pragma unroll
    for (int e = 0; e < 3; ++e) { v[e] = src[tid + 256 * e]; s += v[e] * v[e]; }
    red[tid] = s; __syncthreads();
    for (int off = 128; off > 0; off >>= 1) { if (tid < off) red[tid] += red[tid + off]; __syncthreads(); }
    float inv = 1.0f / fmaxf(sqrtf(red[0]), 1e-12f);
    __bf16* dst = xn + (long)row * CDIM;
#pragma unroll
    for (int e = 0; e < 3; ++e) dst[tid + 256 * e] = f2bf(v[e] * inv);
}

__global__ __launch_bounds__(256) void rowmax_kernel(const float* __restrict__ sim,
                                                     float* __restrict__ maxsim) {
    __shared__ float red[256];
    int row = blockIdx.x;                 // 0 .. B*576-1
    int b = row / NIMG, t = row - b * NIMG;
    int tid = threadIdx.x;
    const float* sr = sim + (long)b * NIMG * NIMG + (long)t * NIMG;
    float m = -3e38f;
    for (int c = tid; c < NIMG; c += 256) m = fmaxf(m, sr[c]);
    red[tid] = m; __syncthreads();
    for (int off = 128; off > 0; off >>= 1) { if (tid < off) red[tid] = fmaxf(red[tid], red[tid + off]); __syncthreads(); }
    if (tid == 0) maxsim[row] = red[0];
}

__global__ __launch_bounds__(576) void select_kernel(const float* __restrict__ maxsim,
                                                     int* __restrict__ keep_idx,
                                                     int* __restrict__ merge_flag) {
    __shared__ float sv[NIMG];
    __shared__ int   kf[NIMG];
    int b = blockIdx.x, i = threadIdx.x;
    sv[i] = maxsim[b * NIMG + i];
    __syncthreads();
    float vi = sv[i];
    int rank = 0;
    for (int j = 0; j < NIMG; ++j) {
        float vj = sv[j];
        rank += (vj > vi) || (vj == vi && j < i);
    }
    int merged = (rank < RKEEP) ? 1 : 0;    // top-r largest max_sim get merged
    kf[i] = 1 - merged;
    merge_flag[b * NIMG + i] = merged;
    __syncthreads();
    if (!merged) {
        int pos = 0;
        for (int j = 0; j < i; ++j) pos += kf[j];
        keep_idx[b * RKEEP + pos] = i;      // sorted ascending by construction
    }
}

__global__ __launch_bounds__(256) void merge_avg_kernel(const float* __restrict__ x2,
                                                        const int* __restrict__ merge_flag,
                                                        float* __restrict__ addvec) {
    int b = blockIdx.x;
    int c = blockIdx.y * 256 + threadIdx.x;
    float s = 0.f;
    const int* mf = merge_flag + b * NIMG;
    for (int t = 0; t < NIMG; ++t)
        if (mf[t]) s += x2[((long)b * SEQ + 1 + t) * CDIM + c];
    addvec[(long)b * CDIM + c] = s * (1.0f / ((float)RKEEP * (float)RKEEP));
}

__global__ __launch_bounds__(256) void gather_out_kernel(const float* __restrict__ x2,
                                                         const int* __restrict__ keep_idx,
                                                         const float* __restrict__ addvec,
                                                         float* __restrict__ out) {
    int row = blockIdx.x;                 // 0 .. B*289-1
    int b = row / (RKEEP + 1), p = row - b * (RKEEP + 1);
    int c = blockIdx.y * 256 + threadIdx.x;
    float val;
    if (p == 0) {
        val = x2[(long)b * SEQ * CDIM + c];
    } else {
        int tok = keep_idx[b * RKEEP + p - 1];
        val = x2[((long)b * SEQ + 1 + tok) * CDIM + c] + addvec[(long)b * CDIM + c];
    }
    out[(long)row * CDIM + c] = val;
}

// ---------------- launcher ----------------
static inline size_t alup(size_t x) { return (x + 255) & ~(size_t)255; }

extern "C" void kernel_launch(void* const* d_in, const int* in_sizes, int n_in,
                              void* d_out, int out_size, void* d_ws, size_t ws_size,
                              hipStream_t stream) {
    const float* x      = (const float*)d_in[0];
    const float* qkv_w  = (const float*)d_in[1];
    const float* qkv_b  = (const float*)d_in[2];
    const float* proj_w = (const float*)d_in[3];
    const float* proj_b = (const float*)d_in[4];
    const float* ln1_g  = (const float*)d_in[5];
    const float* ln1_b  = (const float*)d_in[6];
    const float* ln2_g  = (const float*)d_in[7];
    const float* ln2_b  = (const float*)d_in[8];
    const float* fc1_w  = (const float*)d_in[9];
    const float* fc1_b  = (const float*)d_in[10];
    const float* fc2_w  = (const float*)d_in[11];
    const float* fc2_b  = (const float*)d_in[12];
    float* out = (float*)d_out;

    // workspace layout (bytes); A-side activations padded to MPAD rows for unguarded async loads
    const long WQ_E = (long)3 * CDIM * CDIM;
    const long WP_E = (long)CDIM * CDIM;
    const long W1_E = (long)HID * CDIM;
    const long W2_E = (long)CDIM * HID;
    const long HP_E = (long)MPAD * CDIM;            // padded activation rows
    const long QKV_E = (long)BATCH * HEADS * SPAD * DH;
    const long FC1P_E = (long)MPAD * HID;
    const long SIM_E = (long)BATCH * NIMG * NIMG;
    const long XNP_E = (long)(BATCH * NIMG + 64) * CDIM;  // +64 rows tail pad

    char* base = (char*)d_ws;
    size_t off = 0;
    __bf16* wq_bf  = (__bf16*)(base + off); off = alup(off + WQ_E * 2);
    __bf16* wp_bf  = (__bf16*)(base + off); off = alup(off + WP_E * 2);
    __bf16* w1_bf  = (__bf16*)(base + off); off = alup(off + W1_E * 2);
    __bf16* w2_bf  = (__bf16*)(base + off); off = alup(off + W2_E * 2);
    __bf16* h_bf   = (__bf16*)(base + off); off = alup(off + HP_E * 2);
    size_t qkv_off = off;
    __bf16* qb     = (__bf16*)(base + off); off = alup(off + QKV_E * 2);
    __bf16* kb     = (__bf16*)(base + off); off = alup(off + QKV_E * 2);
    __bf16* vtb    = (__bf16*)(base + off); off = alup(off + QKV_E * 2);
    size_t qkv_bytes = off - qkv_off;
    __bf16* att_bf = (__bf16*)(base + off); off = alup(off + HP_E * 2);
    float*  x1     = (float*)(base + off);  off = alup(off + (long)MROWS * CDIM * 4);
    size_t fc1_off = off;
    __bf16* fc1_bf = (__bf16*)(base + off); off = alup(off + FC1P_E * 2);
    // sim & xn reuse the fc1 region (fc1 activations dead by then)
    float*  sim    = (float*)(base + fc1_off);
    __bf16* xn     = (__bf16*)(base + fc1_off + alup(SIM_E * 4));
    float*  maxsim = (float*)(base + off);  off = alup(off + (long)BATCH * NIMG * 4);
    int*    keepix = (int*)(base + off);    off = alup(off + (long)BATCH * RKEEP * 4);
    int*    mflag  = (int*)(base + off);    off = alup(off + (long)BATCH * NIMG * 4);
    float*  addvec = (float*)(base + off);  off = alup(off + (long)BATCH * CDIM * 4);
    if (off > ws_size) return;  // insufficient scratch; nothing safe to do
    if (alup(SIM_E * 4) + alup(XNP_E * 2) > alup(FC1P_E * 2)) return;

    // 0) zero q/k/vt (covers the SPAD padding)
    hipMemsetAsync((void*)qb, 0, qkv_bytes, stream);

    // 1) weights -> bf16
    cast_bf16_kernel<<<dim3((WQ_E + 255) / 256), 256, 0, stream>>>(qkv_w, wq_bf, WQ_E);
    cast_bf16_kernel<<<dim3((WP_E + 255) / 256), 256, 0, stream>>>(proj_w, wp_bf, WP_E);
    cast_bf16_kernel<<<dim3((W1_E + 255) / 256), 256, 0, stream>>>(fc1_w, w1_bf, W1_E);
    cast_bf16_kernel<<<dim3((W2_E + 255) / 256), 256, 0, stream>>>(fc2_w, w2_bf, W2_E);

    // 2) LN1
    ln_bf16_kernel<<<dim3(MROWS), 256, 0, stream>>>(x, ln1_g, ln1_b, h_bf, MROWS);

    // 3) QKV GEMM -> q/k bf16 [B,H,SPAD,64], V transposed [B,H,64,SPAD]
    gemm_nt_kernel<EPI_QKV><<<dim3(3 * CDIM / 128, MPAD / 128, 1), 256, 0, stream>>>(
        h_bf, wq_bf, qkv_b, MROWS, 3 * CDIM, CDIM, 0, 0, 0,
        nullptr, nullptr, nullptr, qb, kb, vtb);

    // 4) flash attention
    attn_flash_kernel<<<dim3(SPAD / 64, HEADS, BATCH), 128, 0, stream>>>(qb, kb, vtb, att_bf);

    // 5) proj GEMM + residual -> x1 (fp32)
    gemm_nt_kernel<EPI_RES><<<dim3(CDIM / 128, MPAD / 128, 1), 256, 0, stream>>>(
        att_bf, wp_bf, proj_b, MROWS, CDIM, CDIM, 0, 0, 0,
        x1, nullptr, x, nullptr, nullptr, nullptr);

    // 6) LN2 (reuse h_bf)
    ln_bf16_kernel<<<dim3(MROWS), 256, 0, stream>>>(x1, ln2_g, ln2_b, h_bf, MROWS);

    // 7) FC1 GEMM + GELU -> fc1_bf
    gemm_nt_kernel<EPI_GELU><<<dim3(HID / 128, MPAD / 128, 1), 256, 0, stream>>>(
        h_bf, w1_bf, fc1_b, MROWS, HID, CDIM, 0, 0, 0,
        nullptr, fc1_bf, nullptr, nullptr, nullptr, nullptr);

    // 8) FC2 GEMM + residual, in-place into x1 (x2 := x1)
    gemm_nt_kernel<EPI_RES><<<dim3(CDIM / 128, MPAD / 128, 1), 256, 0, stream>>>(
        fc1_bf, w2_bf, fc2_b, MROWS, CDIM, HID, 0, 0, 0,
        x1, nullptr, x1, nullptr, nullptr, nullptr);

    // 9) normalize image tokens -> xn bf16
    norm_rows_kernel<<<dim3(BATCH * NIMG), 256, 0, stream>>>(x1, xn);

    // 10) batched similarity GEMM (per batch 576x576, K=768), diag masked
    gemm_nt_kernel<EPI_SIM><<<dim3((NIMG + 127) / 128, (NIMG + 127) / 128, BATCH), 256, 0, stream>>>(
        xn, xn, nullptr, NIMG, NIMG, CDIM,
        (long)NIMG * CDIM, (long)NIMG * CDIM, (long)NIMG * NIMG,
        sim, nullptr, nullptr, nullptr, nullptr, nullptr);

    // 11) row max, 12) bipartite top-k select, 13) merge average, 14) gather output
    rowmax_kernel<<<dim3(BATCH * NIMG), 256, 0, stream>>>(sim, maxsim);
    select_kernel<<<dim3(BATCH), NIMG, 0, stream>>>(maxsim, keepix, mflag);
    merge_avg_kernel<<<dim3(BATCH, 3), 256, 0, stream>>>(x1, mflag, addvec);
    gather_out_kernel<<<dim3(BATCH * (RKEEP + 1), 3), 256, 0, stream>>>(x1, keepix, addvec, out);
}
